// ConvLSTM_Model_58506044506702
// MI455X (gfx1250) — matrix operand order repaired and checked by reading, not compiled, not run on hardware
//
#include <hip/hip_runtime.h>
#include <math.h>

// ---------------- model constants ----------------
#define L 4
#define NHID 128
#define FC 16
#define HP 16
#define WP 16
#define BB 16
#define T_TOT 20
#define PRE 10
#define NSTEP 19          // T-1
#define GATES (4*NHID)    // 512
#define SP (HP*WP)        // 256
#define HSLAB ((size_t)BB*SP*NHID)          // one layer's h (f16) / c (f32), elements
#define OUT_ELEMS ((size_t)BB*NSTEP*SP*FC)  // 1,245,184

typedef _Float16 h16;
typedef __attribute__((ext_vector_type(16))) _Float16 v16h;
typedef __attribute__((ext_vector_type(8)))  _Float16 v8h;
typedef __attribute__((ext_vector_type(8)))  float    v8f;
typedef __attribute__((ext_vector_type(4)))  unsigned int u32x4;
typedef __attribute__((ext_vector_type(8)))  unsigned int u32x8;

__device__ __forceinline__ float sigm(float x) { return 1.0f / (1.0f + expf(-x)); }

// ---------------- TDM: 1-D contiguous tensor_load_to_lds ----------------
// Copies `units` 8-byte elements from gaddr -> LDS byte offset lds_off.
// Wave-uniform; tracked by TENSORcnt.
__device__ __forceinline__ void tdm_load_1d(unsigned lds_off, unsigned long long gaddr,
                                            unsigned units) {
    u32x4 g0;
    g0.x = 1u;                                   // count=1, user descriptor
    g0.y = lds_off;                              // lds_addr [63:32]
    g0.z = (unsigned)gaddr;                      // global_addr lo
    g0.w = (unsigned)(gaddr >> 32) | (2u << 30); // global_addr hi + type=2 ("image")
    u32x8 g1;
    g1[0] = (3u << 16);                          // workgroup_mask=0, data_size=3 (8B)
    g1[1] = (units & 0xFFFFu) << 16;             // tensor_dim0 lo16 @ bits[63:48]
    g1[2] = (units >> 16) | (1u << 16);          // tensor_dim0 hi | tensor_dim1 lo (=1)
    g1[3] = (units & 0xFFFFu) << 16;             // tensor_dim1 hi (=0) | tile_dim0
    g1[4] = 1u;                                  // tile_dim1=1, tile_dim2=0
    g1[5] = units;                               // tensor_dim0_stride lo
    g1[6] = 0u;
    g1[7] = 0u;
    u32x4 g2 = (u32x4){0u, 0u, 0u, 0u};
    u32x4 g3 = (u32x4){0u, 0u, 0u, 0u};
    asm volatile("tensor_load_to_lds %0, %1, %2, %3"
                 :: "s"(g0), "s"(g1), "s"(g2), "s"(g3)
                 : "memory");
}

// ---------------- weight prep: OIHW f32 -> fragment-packed f16 ----------------
// Layout: [mtile(32)][tap(25)][chunk(nch)][lane(32)][e(16)]  -- the exact per-lane
// WMMA A-operand register image, so a wave's A-load is one coalesced 1KB burst.
__global__ void prep_weights_packed(const float* __restrict__ W, h16* __restrict__ Wt,
                                    int Cin, int nch, int total) {
    int idx = blockIdx.x * blockDim.x + threadIdx.x;
    if (idx >= total) return;
    int e    = idx & 15;
    int lane = (idx >> 4) & 31;
    int rest = idx >> 9;
    int chunk = rest % nch;
    int r2    = rest / nch;
    int tap   = r2 % 25;
    int mtile = r2 / 25;
    int a8 = (lane >> 4) * 16;                      // contiguous-K layout
    int K  = a8 + e;
    int c  = chunk * 32 + K;
    int m  = mtile * 16 + (lane & 15);
    float v = (c < Cin) ? W[((size_t)m * Cin + c) * 25 + tap] : 0.0f;
    Wt[idx] = (h16)v;
}

// ---------------- scheduled-sampling input -> channels-last f16, padded to 32 ch ----
__global__ void make_input(const float* __restrict__ frames, const float* __restrict__ mask,
                           const float* __restrict__ xgen, h16* __restrict__ x, int t) {
    int idx = blockIdx.x * blockDim.x + threadIdx.x;    // 16*256*16 = 65536
    int c = idx & 15, sp = (idx >> 4) & 255, b = idx >> 12;
    float f = frames[((size_t)(b * T_TOT + t) * SP + sp) * FC + c];
    float m = 1.0f;
    if (t >= PRE) m = mask[((size_t)(b * (NSTEP - PRE) + (t - PRE)) * SP + sp) * FC + c];
    float xg = xgen[((size_t)b * SP + sp) * FC + c];
    x[((size_t)b * SP + sp) * 32 + c] = (h16)(m * f + (1.0f - m) * xg);
    // channels 16..31 stay zero from the launch-time memset (never overwritten)
}

// ---------------- implicit-GEMM conv producing 512 gate channels ----------------
// grid.x = 256: bid -> (b, y-pair, m-half). Block = 256 threads (8 waves).
// Each wave computes a 2x2 macro-tile: 2 M-tiles x 2 output rows, so each A-fragment
// feeds 2 WMMAs and each B-fragment feeds 2 WMMAs. LDS slab: [dy(6)][xp(20)][c],
// halo columns/rows zeroed with DS stores, valid rows copied by the TDM (one
// contiguous tensor_load_to_lds per row).
template <int CXP>
__global__ __launch_bounds__(256)
void conv_gates_wmma(const h16* __restrict__ xin,
                     const h16* __restrict__ hin,
                     const h16* __restrict__ WxtP, const h16* __restrict__ WhtP,
                     float* __restrict__ gates) {
    constexpr int NXC = CXP / 32;                 // K-chunks in the x part
    constexpr int XS  = 6 * 20 * CXP;             // x slab size (halves)
    constexpr int HS  = 6 * 20 * NHID;
    __shared__ h16 lds[XS + HS];                  // CXP=128: 61,440 B

    const int bid = blockIdx.x;
    const int b   = bid >> 4;
    const int yp  = (bid >> 1) & 7;
    const int mh  = bid & 1;
    const int y0  = yp * 2;
    const int tid = threadIdx.x;

    // valid slab rows: dy in [dylo, dyhi) maps to image row yy = y0-2+dy
    const int dylo = (2 - y0 > 0) ? (2 - y0) : 0;
    const int dyhi = (18 - y0 < 6) ? (18 - y0) : 6;
    const int ninv = 6 - (dyhi - dylo);

    // zero halo columns (xp = 0,1,18,19) of every row, both tensors
    for (int i = tid; i < 6 * 4 * CXP; i += 256) {
        int c = i % CXP, j = i / CXP;
        int xp4 = j & 3, dy = j >> 2;
        int xp = (xp4 < 2) ? xp4 : (xp4 + 16);
        lds[(dy * 20 + xp) * CXP + c] = (h16)0;
    }
    for (int i = tid; i < 6 * 4 * NHID; i += 256) {
        int c = i % NHID, j = i / NHID;
        int xp4 = j & 3, dy = j >> 2;
        int xp = (xp4 < 2) ? xp4 : (xp4 + 16);
        lds[XS + (dy * 20 + xp) * NHID + c] = (h16)0;
    }
    // zero invalid (clipped) rows entirely
    for (int i = tid; i < ninv * 20 * CXP; i += 256) {
        int c = i % (20 * CXP), k = i / (20 * CXP);
        int dy = (k < dylo) ? k : (dyhi + (k - dylo));
        lds[dy * 20 * CXP + c] = (h16)0;
    }
    for (int i = tid; i < ninv * 20 * NHID; i += 256) {
        int c = i % (20 * NHID), k = i / (20 * NHID);
        int dy = (k < dylo) ? k : (dyhi + (k - dylo));
        lds[XS + dy * 20 * NHID + c] = (h16)0;
    }

    // TDM: one contiguous row copy per valid slab row (dest at xp=2, 16*C halves)
    if (tid < 32) {
        unsigned ldsbase = (unsigned)(size_t)(&lds[0]);
        for (int dy = dylo; dy < dyhi; ++dy) {
            int yy = y0 - 2 + dy;
            unsigned long long gx =
                (unsigned long long)(size_t)(xin + ((size_t)b * SP + yy * 16) * CXP);
            unsigned long long gh =
                (unsigned long long)(size_t)(hin + ((size_t)b * SP + yy * 16) * NHID);
            tdm_load_1d(ldsbase + (unsigned)((dy * 20 + 2) * CXP * 2), gx, (16 * CXP * 2) >> 3);
            tdm_load_1d(ldsbase + (unsigned)((XS + (dy * 20 + 2) * NHID) * 2), gh,
                        (16 * NHID * 2) >> 3);
        }
        __builtin_amdgcn_s_wait_tensorcnt(0);
    }
    __syncthreads();

    const int wave = tid >> 5;
    const int lane = tid & 31;
    const int ln   = lane & 15;          // N column (x) / M row-in-tile
    const int hi4  = lane >> 4;
    const int b16  = hi4 * 16;           // B-fragment K base

    v8f acc[2][2];                       // [mi][row]
#pragma unroll
    for (int mi = 0; mi < 2; ++mi)
#pragma unroll
        for (int r = 0; r < 2; ++r)
            acc[mi][r] = (v8f){0.f,0.f,0.f,0.f,0.f,0.f,0.f,0.f};

    const int mt0 = mh * 16 + wave * 2;  // first of this wave's 2 M-tiles

#pragma unroll 1
    for (int tap = 0; tap < 25; ++tap) {
        const int ty = tap / 5, tx = tap % 5;
        const int xp = ln + tx;          // 0..19, padded layout: no bounds check

#pragma unroll
        for (int ch = 0; ch < NXC; ++ch) {
            v16h bfrag[2];
#pragma unroll
            for (int r = 0; r < 2; ++r)
                bfrag[r] = *(const v16h*)(&lds[((ty + r) * 20 + xp) * CXP + ch * 32 + b16]);
            v16h afrag[2];
#pragma unroll
            for (int mi = 0; mi < 2; ++mi)
                afrag[mi] = *(const v16h*)(WxtP +
                    (((size_t)((mt0 + mi) * 25 + tap) * NXC + ch) << 9) + (lane << 4));
#pragma unroll
            for (int mi = 0; mi < 2; ++mi)
#pragma unroll
                for (int r = 0; r < 2; ++r)
                    acc[mi][r] = __builtin_amdgcn_wmma_f32_16x16x32_f16(
                        false, afrag[mi], false, bfrag[r], (short)0, acc[mi][r], false, false);
        }
#pragma unroll
        for (int ch = 0; ch < 4; ++ch) {
            v16h bfrag[2];
#pragma unroll
            for (int r = 0; r < 2; ++r)
                bfrag[r] = *(const v16h*)(&lds[XS + ((ty + r) * 20 + xp) * NHID + ch * 32 + b16]);
            v16h afrag[2];
#pragma unroll
            for (int mi = 0; mi < 2; ++mi)
                afrag[mi] = *(const v16h*)(WhtP +
                    (((size_t)((mt0 + mi) * 25 + tap) * 4 + ch) << 9) + (lane << 4));
#pragma unroll
            for (int mi = 0; mi < 2; ++mi)
#pragma unroll
                for (int r = 0; r < 2; ++r)
                    acc[mi][r] = __builtin_amdgcn_wmma_f32_16x16x32_f16(
                        false, afrag[mi], false, bfrag[r], (short)0, acc[mi][r], false, false);
        }
    }

    // epilogue: channels-last gates[b][y][x][m]; 32B contiguous per accumulator
#pragma unroll
    for (int r = 0; r < 2; ++r) {
        float* gp = gates + (((size_t)b * SP) + (y0 + r) * 16 + ln) * GATES;
#pragma unroll
        for (int mi = 0; mi < 2; ++mi)
            *(v8f*)(gp + (mt0 + mi) * 16 + hi4 * 8) = acc[mi][r];
    }
}

// ---------------- LSTM gate nonlinearity; h written as f16 channels-last --------
__global__ void lstm_gates(const float* __restrict__ gates,
                           h16* __restrict__ h, float* __restrict__ c) {
    int idx = blockIdx.x * blockDim.x + threadIdx.x;   // 16*256*128 = 524288
    int ch = idx & 127;
    int sp = (idx >> 7) & 255;
    int b  = idx >> 15;
    const float* g = gates + ((size_t)b * SP + sp) * GATES;
    float ig = g[ch];
    float fg = g[128 + ch];
    float gg = g[256 + ch];
    float og = g[384 + ch];
    size_t o = ((size_t)b * SP + sp) * NHID + ch;
    float cn = sigm(fg + 1.0f) * c[o] + sigm(ig) * tanhf(gg);
    float hn = sigm(og) * tanhf(cn);
    c[o] = cn;
    h[o] = (h16)hn;
}

// ---------------- 1x1 projection -> x_gen (f32 CL) and output frame ----------------
__global__ void last_conv(const h16* __restrict__ h3, const float* __restrict__ Wl,
                          float* __restrict__ xgen, float* __restrict__ out, int t) {
    int idx = blockIdx.x * blockDim.x + threadIdx.x;   // 65536
    int fc = idx & 15, sp = (idx >> 4) & 255, b = idx >> 12;
    const h16* hp = h3 + ((size_t)b * SP + sp) * NHID;
    float s = 0.0f;
#pragma unroll 8
    for (int cc = 0; cc < NHID; ++cc) s += Wl[fc * NHID + cc] * (float)hp[cc];
    xgen[((size_t)b * SP + sp) * FC + fc] = s;
    out[((size_t)(b * NSTEP + t) * SP + sp) * FC + fc] = s;
}

// ---------------- loss: MSE part ----------------
__global__ void mse_kernel(const float* __restrict__ out, const float* __restrict__ frames,
                           float* __restrict__ accum) {
    __shared__ float red[256];
    size_t idx = (size_t)blockIdx.x * 256 + threadIdx.x;
    float d = 0.0f;
    if (idx < OUT_ELEMS) {
        int cc = idx & 15;
        size_t r = idx >> 4;
        int sp = r & 255;
        size_t q = r >> 8;
        int t = (int)(q % NSTEP);
        int b = (int)(q / NSTEP);
        float p  = out[idx];
        float tr = frames[(((size_t)b * T_TOT + (t + 1)) * SP + sp) * FC + cc];
        d = (p - tr) * (p - tr);
    }
    red[threadIdx.x] = d;
    __syncthreads();
    for (int s = 128; s > 0; s >>= 1) {
        if (threadIdx.x < s) red[threadIdx.x] += red[threadIdx.x + s];
        __syncthreads();
    }
    if (threadIdx.x == 0) atomicAdd(accum, red[0]);
}

// ---------------- loss: xi heuristic (softmax over batch dim) ----------------
__global__ void xi_kernel(const float* __restrict__ a, int is_pred, float* __restrict__ xq) {
    int idx = blockIdx.x * blockDim.x + threadIdx.x;   // 16*16*19 = 4864
    if (idx >= FC * WP * NSTEP) return;
    int t = idx % NSTEP;
    int w = (idx / NSTEP) % WP;
    int c = idx / (NSTEP * WP);
    float v[BB];
    float mx = -1e30f;
    for (int b = 0; b < BB; ++b) {
        float p = is_pred
            ? a[(((size_t)b * NSTEP + t) * SP + w) * FC + c]                // h==0 row
            : a[(((size_t)b * T_TOT + (t + 1)) * SP + w) * FC + c];
        v[b] = p;
        mx = fmaxf(mx, p);
    }
    float se = 0.0f, ws = 0.0f;
    for (int b = 0; b < BB; ++b) {
        float e = expf(v[b] - mx);
        se += e;
        ws += e * (float)b;
    }
    atomicAdd(&xq[w * NSTEP + t], (ws / se) * (1.0f / (float)FC));
}

__global__ void finalize_loss(const float* __restrict__ mse, const float* __restrict__ xqp,
                              const float* __restrict__ xqt, float* __restrict__ loss_out) {
    if (threadIdx.x == 0 && blockIdx.x == 0) {
        float s = 0.0f;
        for (int i = 0; i < WP * NSTEP; ++i) {
            float d = xqp[i] - xqt[i];
            s += d * d;
        }
        loss_out[0] = mse[0] / (float)OUT_ELEMS + s / (float)(WP * NSTEP);
    }
}

// ---------------- host driver ----------------
extern "C" void kernel_launch(void* const* d_in, const int* in_sizes, int n_in,
                              void* d_out, int out_size, void* d_ws, size_t ws_size,
                              hipStream_t stream) {
    (void)in_sizes; (void)n_in; (void)out_size; (void)ws_size;
    const float* frames = (const float*)d_in[0];
    const float* mask   = (const float*)d_in[1];
    const float* Wx[L]  = {(const float*)d_in[2], (const float*)d_in[4],
                           (const float*)d_in[6], (const float*)d_in[8]};
    const float* Wh[L]  = {(const float*)d_in[3], (const float*)d_in[5],
                           (const float*)d_in[7], (const float*)d_in[9]};
    const float* Wl     = (const float*)d_in[10];
    float* out = (float*)d_out;

    // workspace carve-up (256B-aligned)
    char* ws = (char*)d_ws;
    size_t off = 0;
    auto carve = [&](size_t bytes) -> void* {
        void* p = ws + off;
        off += (bytes + 255) & ~(size_t)255;
        return p;
    };
    h16* Wxt[L];
    h16* Wht[L];
    Wxt[0] = (h16*)carve((size_t)32 * 25 * 1 * 512 * sizeof(h16));
    Wht[0] = (h16*)carve((size_t)32 * 25 * 4 * 512 * sizeof(h16));
    for (int i = 1; i < L; ++i) {
        Wxt[i] = (h16*)carve((size_t)32 * 25 * 4 * 512 * sizeof(h16));
        Wht[i] = (h16*)carve((size_t)32 * 25 * 4 * 512 * sizeof(h16));
    }
    h16*   hbuf  = (h16*)carve(L * HSLAB * sizeof(h16));     // f16 channels-last
    float* cbuf  = (float*)carve(L * HSLAB * sizeof(float)); // f32 channels-last
    float* gates = (float*)carve((size_t)BB * SP * GATES * sizeof(float));
    h16*   xbuf  = (h16*)carve((size_t)BB * SP * 32 * sizeof(h16));
    float* xgen  = (float*)carve((size_t)BB * SP * FC * sizeof(float));
    float* accum = (float*)carve(4096);
    float* xqp = accum + 16;
    float* xqt = xqp + WP * NSTEP;

    hipMemsetAsync(hbuf, 0, L * HSLAB * sizeof(h16), stream);
    hipMemsetAsync(cbuf, 0, L * HSLAB * sizeof(float), stream);
    hipMemsetAsync(xbuf, 0, (size_t)BB * SP * 32 * sizeof(h16), stream);  // pad channels
    hipMemsetAsync(xgen, 0, (size_t)BB * SP * FC * sizeof(float), stream);
    hipMemsetAsync(accum, 0, 4096, stream);

    // one-time weight repack to fragment order (constant across all 19 steps)
    {
        int tot = 32 * 25 * 1 * 512;
        prep_weights_packed<<<(tot + 255) / 256, 256, 0, stream>>>(Wx[0], Wxt[0], FC, 1, tot);
        tot = 32 * 25 * 4 * 512;
        prep_weights_packed<<<(tot + 255) / 256, 256, 0, stream>>>(Wh[0], Wht[0], NHID, 4, tot);
        for (int i = 1; i < L; ++i) {
            prep_weights_packed<<<(tot + 255) / 256, 256, 0, stream>>>(Wx[i], Wxt[i], NHID, 4, tot);
            prep_weights_packed<<<(tot + 255) / 256, 256, 0, stream>>>(Wh[i], Wht[i], NHID, 4, tot);
        }
    }

    for (int t = 0; t < NSTEP; ++t) {
        make_input<<<256, 256, 0, stream>>>(frames, mask, xgen, xbuf, t);
        for (int i = 0; i < L; ++i) {
            const h16* xin = (i == 0) ? xbuf : (hbuf + (size_t)(i - 1) * HSLAB);
            h16*   hi = hbuf + (size_t)i * HSLAB;
            float* ci = cbuf + (size_t)i * HSLAB;
            if (i == 0)
                conv_gates_wmma<32><<<256, 256, 0, stream>>>(xin, hi, Wxt[i], Wht[i], gates);
            else
                conv_gates_wmma<128><<<256, 256, 0, stream>>>(xin, hi, Wxt[i], Wht[i], gates);
            lstm_gates<<<(int)((BB * SP * NHID) / 256), 256, 0, stream>>>(gates, hi, ci);
        }
        last_conv<<<256, 256, 0, stream>>>(hbuf + 3 * HSLAB, Wl, xgen, out, t);
    }

    // loss
    mse_kernel<<<(unsigned)(OUT_ELEMS / 256), 256, 0, stream>>>(out, frames, accum);
    xi_kernel<<<NSTEP, 256, 0, stream>>>(out, 1, xqp);
    xi_kernel<<<NSTEP, 256, 0, stream>>>(frames, 0, xqt);
    finalize_loss<<<1, 1, 0, stream>>>(accum, xqp, xqt, out + OUT_ELEMS);
}
